// GraphSimilarity_56573309223901
// MI455X (gfx1250) — compile-verified, matrix operand order checked by reading, not run on hardware
//
#include <hip/hip_runtime.h>
#include <hip/hip_bf16.h>

typedef __attribute__((ext_vector_type(16))) __bf16 v16bf;
typedef __attribute__((ext_vector_type(8)))  float  v8f;

#define HDIM 128
#define LDS_PITCH 132   // 132 % 64 = 4 -> conflict-free column reads of the A tile

// Fragment K index for 16-bit WMMA A/B operands (ISA 16x32 bf16 layout):
// half h (0..15), lane-half hi (0/1): K = (vg>=4?16:0) + 8*hi + 2*(vg&3) + (h&1)
__device__ __forceinline__ int frag_k(int h, int hi) {
  const int vg = h >> 1;
  return ((vg & 4) << 2) + (hi << 3) + ((vg & 3) << 1) + (h & 1);
}

// ---------------------------------------------------------------------------
// One-time weight repack: f32 [128,128] row-major (K,N) -> bf16 in exact WMMA
// B-fragment order: [ntile(8)][kstep(4)][lane(32)][half(16)].  32 KB per weight.
// ---------------------------------------------------------------------------
__global__ __launch_bounds__(256) void pack_weight(
    const float* __restrict__ W, __bf16* __restrict__ Wp) {
  const int i = blockIdx.x * 256 + threadIdx.x;   // 0..16383
  const int h = i & 15;
  const int l = (i >> 4) & 31;
  const int s = (i >> 9) & 3;
  const int t = i >> 11;
  const int k = frag_k(h, (l >> 4) & 1);
  Wp[i] = (__bf16)W[(s * 32 + k) * HDIM + t * 16 + (l & 15)];
}

// ---------------------------------------------------------------------------
// GEMM: out[N,128] = (relu_in? relu(A) : A) @ W + bias
// block = 256 threads (8 waves); block owns a 16-row M tile, wave w owns cols
// [16w,16w+16).  A staged f32 in LDS (padded pitch), B read as packed bf16
// fragments (2x global_load_b128 per k-step).  4x v_wmma_f32_16x16x32_bf16.
// ---------------------------------------------------------------------------
__global__ __launch_bounds__(256) void gemm128_wmma_bf16(
    const float* __restrict__ A, const __bf16* __restrict__ Wp,
    const float* __restrict__ bias, float* __restrict__ out,
    int n_rows, int relu_in) {
  __shared__ float As[16 * LDS_PITCH];

  const int tid  = threadIdx.x;
  const int wave = tid >> 5;
  const int lane = tid & 31;
  const long row0 = (long)blockIdx.x * 16;

  // cooperative, coalesced stage of the 16x128 A tile (fused input ReLU)
  for (int i = tid; i < 16 * HDIM; i += 256) {
    int r = i >> 7, c = i & 127;
    float v = (row0 + r < n_rows) ? A[(row0 + r) * HDIM + c] : 0.0f;
    if (relu_in) v = fmaxf(v, 0.0f);
    As[r * LDS_PITCH + c] = v;
  }
  __syncthreads();

  const int m  = lane & 15;
  const int hi = lane >> 4;
  const int n  = (wave << 4) + m;
  const __bf16* wp_lane = Wp + ((wave * 4) * 32 + lane) * 16;

  v8f acc = {};
#pragma unroll
  for (int s = 0; s < 4; ++s) {
    v16bf a = {};
#pragma unroll
    for (int h = 0; h < 16; ++h)
      a[h] = (__bf16)As[m * LDS_PITCH + s * 32 + frag_k(h, hi)];
    const v16bf b = *(const v16bf*)(wp_lane + s * 32 * 16);
    acc = __builtin_amdgcn_wmma_f32_16x16x32_bf16(
        /*neg_a=*/false, a, /*neg_b=*/false, b,
        /*c_mod=*/(short)0, acc, /*reuse_a=*/false, /*reuse_b=*/false);
  }

  const float bv = bias[n];
#pragma unroll
  for (int j = 0; j < 8; ++j) {
    const long mm = row0 + j + (hi << 3);   // C/D layout: M = j + 8*hi
    if (mm < n_rows) out[mm * HDIM + n] = acc[j] + bv;
  }
}

// ---------------------------------------------------------------------------
// Degree / normalization
// ---------------------------------------------------------------------------
__global__ void zero_f32(float* __restrict__ p, long n) {
  long i = (long)blockIdx.x * blockDim.x + threadIdx.x;
  if (i < n) p[i] = 0.0f;
}

__global__ void count_deg(const int* __restrict__ rows, float* __restrict__ deg, int E) {
  int e = blockIdx.x * blockDim.x + threadIdx.x;
  if (e < E) atomicAdd(&deg[rows[e]], 1.0f);
}

__global__ void compute_dis(const float* __restrict__ deg, float* __restrict__ dis, int n) {
  int i = blockIdx.x * blockDim.x + threadIdx.x;
  if (i < n) {
    float d = deg[i];
    dis[i] = (d > 0.0f) ? rsqrtf(d) : 0.0f;
  }
}

// ---------------------------------------------------------------------------
// Edge scatter: out[row] += hW[col] * (dis[row]*dis[col]).  One wave per edge,
// each lane handles a float4 (coalesced 512B row read, 4 f32 L2 atomics).
// ---------------------------------------------------------------------------
__global__ __launch_bounds__(256) void scatter_edges(
    const float* __restrict__ hW, const int* __restrict__ rows,
    const int* __restrict__ cols, const float* __restrict__ dis,
    float* __restrict__ out, int E) {
  int e = blockIdx.x * 8 + (threadIdx.x >> 5);
  if (e >= E) return;
  int lane = threadIdx.x & 31;
  int r = rows[e], c = cols[e];
  float norm = dis[r] * dis[c];
  float4 v = ((const float4*)(hW + (long)c * HDIM))[lane];
  float* dst = out + (long)r * HDIM + lane * 4;
  atomicAdd(dst + 0, v.x * norm);
  atomicAdd(dst + 1, v.y * norm);
  atomicAdd(dst + 2, v.z * norm);
  atomicAdd(dst + 3, v.w * norm);
}

// ---------------------------------------------------------------------------
// Pooling: gsum[f] += sum over a 256-row chunk of relu(h[:,f])  (fused ReLU)
// ---------------------------------------------------------------------------
__global__ __launch_bounds__(128) void col_sum_relu(
    const float* __restrict__ h, float* __restrict__ gsum, int n_rows) {
  int f = threadIdx.x;
  long r0 = (long)blockIdx.x * 256;
  float s = 0.0f;
  for (int r = 0; r < 256; ++r) {
    long rr = r0 + r;
    if (rr < n_rows) s += fmaxf(h[rr * HDIM + f], 0.0f);
  }
  atomicAdd(&gsum[f], s);
}

// ---------------------------------------------------------------------------
// Similarity head: c=[g1,g2,|g1-g2|,g1*g2] (512) -> relu(c@W1+b1) (128) -> @W2+b2
// ---------------------------------------------------------------------------
__global__ __launch_bounds__(128) void sim_head(
    const float* __restrict__ gsum, const float* __restrict__ w1,
    const float* __restrict__ b1, const float* __restrict__ w2,
    const float* __restrict__ b2, float* __restrict__ out, float invN) {
  __shared__ float c[512];
  __shared__ float red[128];
  int t = threadIdx.x;
  float g1 = gsum[t] * invN;
  float g2 = gsum[HDIM + t] * invN;
  c[t]       = g1;
  c[128 + t] = g2;
  c[256 + t] = fabsf(g1 - g2);
  c[384 + t] = g1 * g2;
  __syncthreads();
  float acc = b1[t];
  for (int k = 0; k < 512; ++k) acc = fmaf(c[k], w1[k * HDIM + t], acc);
  acc = fmaxf(acc, 0.0f);
  red[t] = acc * w2[t];
  __syncthreads();
  for (int s = 64; s > 0; s >>= 1) {
    if (t < s) red[t] += red[t + s];
    __syncthreads();
  }
  if (t == 0) out[0] = red[0] + b2[0];
}

// ---------------------------------------------------------------------------
extern "C" void kernel_launch(void* const* d_in, const int* in_sizes, int n_in,
                              void* d_out, int out_size, void* d_ws, size_t ws_size,
                              hipStream_t stream) {
  const float* x[2]  = {(const float*)d_in[0], (const float*)d_in[2]};
  const int*   ei[2] = {(const int*)d_in[1],   (const int*)d_in[3]};
  const float* enc_w1 = (const float*)d_in[4];
  const float* enc_b1 = (const float*)d_in[5];
  const float* enc_w2 = (const float*)d_in[6];
  const float* enc_b2 = (const float*)d_in[7];
  const float* conv1_w = (const float*)d_in[8];
  const float* conv1_b = (const float*)d_in[9];
  const float* conv2_w = (const float*)d_in[10];
  const float* conv2_b = (const float*)d_in[11];
  const float* sim_w1 = (const float*)d_in[12];
  const float* sim_b1 = (const float*)d_in[13];
  const float* sim_w2 = (const float*)d_in[14];
  const float* sim_b2 = (const float*)d_in[15];

  const int N = in_sizes[0] / HDIM;
  const int E = in_sizes[1] / 2;
  const long NH = (long)N * HDIM;
  const int WPK = 8 * 4 * 32 * 16;   // 16384 bf16 per packed weight

  // workspace carve-up: packed weights first (keeps 16B alignment), then bufs
  __bf16* wpk0 = (__bf16*)d_ws;           // enc_w1
  __bf16* wpk1 = wpk0 + WPK;              // enc_w2
  __bf16* wpk2 = wpk1 + WPK;              // conv1_w
  __bf16* wpk3 = wpk2 + WPK;              // conv2_w
  float* bufA = (float*)(wpk3 + WPK);
  float* bufB = bufA + NH;
  float* bufC = bufB + NH;
  float* deg  = bufC + NH;
  float* dis  = deg + N;
  float* gsum = dis + N;                  // 2 * 128 floats

  // one-time (per launch) weight repack to WMMA bf16 fragment order
  pack_weight<<<64, 256, 0, stream>>>(enc_w1, wpk0);
  pack_weight<<<64, 256, 0, stream>>>(enc_w2, wpk1);
  pack_weight<<<64, 256, 0, stream>>>(conv1_w, wpk2);
  pack_weight<<<64, 256, 0, stream>>>(conv2_w, wpk3);

  const int gemm_grid = (N + 15) / 16;
  const int zb_NH = (int)((NH + 255) / 256);
  const int zb_N  = (N + 255) / 256;
  const int eb    = (E + 255) / 256;
  const int sb    = (E + 7) / 8;          // 8 edges (waves) per 256-thread block
  const int pb    = (N + 255) / 256;      // 256-row chunks for pooling

  for (int g = 0; g < 2; ++g) {
    const int* rows = ei[g];
    const int* cols = ei[g] + E;

    // deg^-0.5 normalization
    zero_f32<<<zb_N, 256, 0, stream>>>(deg, N);
    count_deg<<<eb, 256, 0, stream>>>(rows, deg, E);
    compute_dis<<<zb_N, 256, 0, stream>>>(deg, dis, N);

    // encoder MLP: h = relu(x@W1+b1)@W2+b2  (ReLU fused into 2nd GEMM's load)
    gemm128_wmma_bf16<<<gemm_grid, 256, 0, stream>>>(x[g],  wpk0, enc_b1, bufA, N, 0);
    gemm128_wmma_bf16<<<gemm_grid, 256, 0, stream>>>(bufA, wpk1, enc_b2, bufB, N, 1);

    // GCN layer 1: hW = h@W+b ; aggregate (ReLU deferred to next GEMM's load)
    gemm128_wmma_bf16<<<gemm_grid, 256, 0, stream>>>(bufB, wpk2, conv1_b, bufA, N, 0);
    zero_f32<<<zb_NH, 256, 0, stream>>>(bufC, NH);
    scatter_edges<<<sb, 256, 0, stream>>>(bufA, rows, cols, dis, bufC, E);

    // GCN layer 2 (reads relu(bufC)); aggregate; ReLU fused into pooling
    gemm128_wmma_bf16<<<gemm_grid, 256, 0, stream>>>(bufC, wpk3, conv2_b, bufA, N, 1);
    zero_f32<<<zb_NH, 256, 0, stream>>>(bufB, NH);
    scatter_edges<<<sb, 256, 0, stream>>>(bufA, rows, cols, dis, bufB, E);

    // mean pooling (sum of relu(h); divide in head)
    zero_f32<<<1, 128, 0, stream>>>(gsum + g * HDIM, HDIM);
    col_sum_relu<<<pb, 128, 0, stream>>>(bufB, gsum + g * HDIM, N);
  }

  sim_head<<<1, 128, 0, stream>>>(gsum, sim_w1, sim_b1, sim_w2, sim_b2,
                                  (float*)d_out, 1.0f / (float)N);
}
